// FixedGemmaMLP_42142219108650
// MI455X (gfx1250) — matrix operand order repaired and checked by reading, not compiled, not run on hardware
//
#include <hip/hip_runtime.h>
#include <hip/hip_bf16.h>
#include <cstdint>

// ---------------------------------------------------------------------------
// FixedGemmaMLP for gfx1250 (MI455X): bf16 WMMA GEMMs with f32 accumulation,
// double-buffered LDS tiles filled with GLOBAL_LOAD_ASYNC_TO_LDS_B128
// (ASYNCcnt-tracked, no VGPR staging).
//   hidden [8192,4096] f32 -> bf16
//   Wg/Wu/Wd [4096,4096] f32 -> bf16, transposed to [N][K]
//   kernel1: gate/up dual-GEMM + bf16-exact GELU -> gated bf16
//   kernel2: gated @ WdT -> out bf16
// ---------------------------------------------------------------------------

#define MTOT 8192
#define KH   4096
#define BK   64
#define NKT  (KH / BK)      // 64 k-tiles
#define LROW 72             // padded LDS row (ushorts): 144B, 16B-aligned steps

typedef unsigned short ushort_t;
typedef __bf16 v16bf __attribute__((ext_vector_type(16)));
typedef float  v8f   __attribute__((ext_vector_type(8)));
typedef unsigned short u16x8 __attribute__((ext_vector_type(8)));

struct FragBits { u16x8 lo, hi; };   // 32 bytes == v16bf

__device__ __forceinline__ ushort_t f2bf(float f) {
  unsigned x = __float_as_uint(f);
  x = x + 0x7fffu + ((x >> 16) & 1u);          // round-to-nearest-even
  return (ushort_t)(x >> 16);
}
__device__ __forceinline__ float bf2f(ushort_t b) {
  return __uint_as_float(((unsigned)b) << 16);
}

// Async global->LDS copy of 16 bytes per lane (CDNA5, ASYNCcnt-tracked).
// LDS operand: low 32 bits of a shared-aperture address == wave LDS offset.
__device__ __forceinline__ void async_cp16(void* lds, const void* gptr) {
  asm volatile("global_load_async_to_lds_b128 %0, %1, off"
               :: "v"((unsigned)(size_t)lds), "v"(gptr)
               : "memory");
}
#define WAIT_ASYNCCNT(n) asm volatile("s_wait_asynccnt " #n ::: "memory")

// --------------------------- conversion kernels -----------------------------

__global__ __launch_bounds__(256) void cvt_f32_to_bf16(
    const float* __restrict__ src, ushort_t* __restrict__ dst) {
  const size_t base = ((size_t)blockIdx.x * 256 + threadIdx.x) * 8;
  const float4* s4 = (const float4*)(src + base);
  float4 a = s4[0], b = s4[1];
  u16x8 o;
  o[0] = f2bf(a.x); o[1] = f2bf(a.y); o[2] = f2bf(a.z); o[3] = f2bf(a.w);
  o[4] = f2bf(b.x); o[5] = f2bf(b.y); o[6] = f2bf(b.z); o[7] = f2bf(b.w);
  *(u16x8*)(dst + base) = o;
}

// W[k][n] f32 -> Wt[n][k] bf16  (32x32 LDS tile transpose, z selects weight)
__global__ __launch_bounds__(256) void cvtT_f32_to_bf16(
    const float* __restrict__ g, const float* __restrict__ u,
    const float* __restrict__ d,
    ushort_t* __restrict__ gw, ushort_t* __restrict__ uw,
    ushort_t* __restrict__ dw) {
  const float* src = (blockIdx.z == 0) ? g : (blockIdx.z == 1) ? u : d;
  ushort_t*    dst = (blockIdx.z == 0) ? gw : (blockIdx.z == 1) ? uw : dw;
  __shared__ ushort_t tile[32][33];
  const int k0 = blockIdx.y * 32, n0 = blockIdx.x * 32;
  const int tx = threadIdx.x, ty = threadIdx.y;
#pragma unroll
  for (int i = 0; i < 4; ++i) {
    const int k = k0 + ty + i * 8;
    tile[ty + i * 8][tx] = f2bf(src[(size_t)k * KH + n0 + tx]);
  }
  __syncthreads();
#pragma unroll
  for (int i = 0; i < 4; ++i) {
    const int n = n0 + ty + i * 8;
    dst[(size_t)n * KH + k0 + tx] = tile[tx][ty + i * 8];
  }
}

// ------------------------------ WMMA helpers --------------------------------
// A frag (16x32 bf16): lane row M = lane&15; K chunks {h*8..+7, h*8+16..+23}.
// B frag (32x16 bf16) from Bt[n][k]: lane col N = lane&15; K chunks {h*8..+7, h*8+16..+23}.

__device__ __forceinline__ v16bf load_a_frag(const ushort_t* rowptr, int half) {
  FragBits fb;
  fb.lo = *(const u16x8*)(rowptr + half * 8);
  fb.hi = *(const u16x8*)(rowptr + half * 8 + 16);
  return __builtin_bit_cast(v16bf, fb);
}
__device__ __forceinline__ v16bf load_b_frag(const ushort_t* rowptr, int half) {
  FragBits fb;
  fb.lo = *(const u16x8*)(rowptr + half * 8);
  fb.hi = *(const u16x8*)(rowptr + half * 8 + 16);
  return __builtin_bit_cast(v16bf, fb);
}

// --------------------- kernel 1: gate/up GEMM + GELU ------------------------
// Block tile 128(M) x 64(N), BK = 64. 8 wave32s; wave tile 32x32 per GEMM.
// Double-buffered LDS, filled by async-to-LDS; 8 async B128 ops per thread
// per tile (A:4, Wg:2, Wu:2).

__global__ __launch_bounds__(256) void mlp_gateup(
    const ushort_t* __restrict__ A,    // [MTOT][KH] bf16
    const ushort_t* __restrict__ Gw,   // [KH(n)][KH(k)] bf16 (transposed)
    const ushort_t* __restrict__ Uw,   // [KH(n)][KH(k)] bf16 (transposed)
    ushort_t* __restrict__ Gated) {    // [MTOT][KH] bf16
  __shared__ __align__(16) ushort_t As[2][128][LROW];
  __shared__ __align__(16) ushort_t Bs[2][2][64][LROW];

  const int tid  = threadIdx.x;
  const int lane = tid & 31;
  const int wave = tid >> 5;
  const int half = lane >> 4;
  const int l16  = lane & 15;
  const int wm   = wave & 3;            // 4 wave rows  * 32 = 128
  const int wn   = wave >> 2;           // 2 wave cols  * 32 = 64
  const int m0   = blockIdx.y * 128;
  const int n0   = blockIdx.x * 64;

  v8f acc[2][2][2] = {};                // [gemm][msub][nsub]

  auto issue = [&](int buf, int k0) {
    // A tile: 128 rows x 64 cols = 1024 x 16B chunks -> 4 per thread
#pragma unroll
    for (int i = 0; i < 4; ++i) {
      const int c = tid + i * 256;
      const int row = c >> 3, seg = (c & 7) * 8;
      async_cp16(&As[buf][row][seg],
                 A + (size_t)(m0 + row) * KH + k0 + seg);
    }
    // B tiles: 64 rows x 64 cols = 512 chunks per weight -> 2 per thread
#pragma unroll
    for (int i = 0; i < 2; ++i) {
      const int c = tid + i * 256;
      const int n = c >> 3, seg = (c & 7) * 8;
      async_cp16(&Bs[buf][0][n][seg],
                 Gw + (size_t)(n0 + n) * KH + k0 + seg);
      async_cp16(&Bs[buf][1][n][seg],
                 Uw + (size_t)(n0 + n) * KH + k0 + seg);
    }
  };

  issue(0, 0);
  int cur = 0;
  for (int t = 0; t < NKT; ++t) {
    if (t + 1 < NKT) {
      issue(cur ^ 1, (t + 1) * BK);
      WAIT_ASYNCCNT(8);      // this tile's 8 ops complete (in-order)
    } else {
      WAIT_ASYNCCNT(0);
    }
    __syncthreads();         // publish LDS tile to all waves

#pragma unroll
    for (int kk = 0; kk < 2; ++kk) {
      v16bf af[2];
#pragma unroll
      for (int ms = 0; ms < 2; ++ms)
        af[ms] = load_a_frag(&As[cur][wm * 32 + ms * 16 + l16][kk * 32], half);
      v16bf bfr[2][2];
#pragma unroll
      for (int g = 0; g < 2; ++g)
#pragma unroll
        for (int ns = 0; ns < 2; ++ns)
          bfr[g][ns] =
              load_b_frag(&Bs[cur][g][wn * 32 + ns * 16 + l16][kk * 32], half);
#pragma unroll
      for (int g = 0; g < 2; ++g)
#pragma unroll
        for (int ms = 0; ms < 2; ++ms)
#pragma unroll
          for (int ns = 0; ns < 2; ++ns)
            acc[g][ms][ns] = __builtin_amdgcn_wmma_f32_16x16x32_bf16(
                false, af[ms], false, bfr[g][ns], (short)0, acc[g][ms][ns],
                false, false);
    }
    __syncthreads();         // done reading before buffer is re-issued
    cur ^= 1;
  }

  // ---- epilogue: bf16-exact GELU(gate) * up -> Gated -----------------------
#pragma unroll
  for (int ms = 0; ms < 2; ++ms)
#pragma unroll
    for (int ns = 0; ns < 2; ++ns) {
#pragma unroll
      for (int r = 0; r < 8; ++r) {
        const int row = m0 + wm * 32 + ms * 16 + half * 8 + r;
        const int col = n0 + wn * 32 + ns * 16 + l16;
        float x     = bf2f(f2bf(acc[0][ms][ns][r]));
        float x3    = bf2f(f2bf(x * x * x));
        float inner = bf2f(f2bf(0.7978845608028654f * (x + 0.044715f * x3)));
        float t     = bf2f(f2bf(tanhf(inner)));
        float act   = bf2f(f2bf(0.5f * x * (1.0f + t)));
        float u     = bf2f(f2bf(acc[1][ms][ns][r]));
        Gated[(size_t)row * KH + col] = f2bf(act * u);
      }
    }
}

// ----------------------- kernel 2: down projection --------------------------
// Same pipeline, single GEMM; 6 async B128 ops per thread per tile (A:4, B:2).

__global__ __launch_bounds__(256) void mlp_down(
    const ushort_t* __restrict__ A,    // gated [MTOT][KH] bf16
    const ushort_t* __restrict__ Dw,   // [KH(n)][KH(k)] bf16 (transposed)
    ushort_t* __restrict__ Out) {      // [MTOT][KH] bf16 bits
  __shared__ __align__(16) ushort_t As[2][128][LROW];
  __shared__ __align__(16) ushort_t Bs[2][64][LROW];

  const int tid  = threadIdx.x;
  const int lane = tid & 31;
  const int wave = tid >> 5;
  const int half = lane >> 4;
  const int l16  = lane & 15;
  const int wm   = wave & 3;
  const int wn   = wave >> 2;
  const int m0   = blockIdx.y * 128;
  const int n0   = blockIdx.x * 64;

  v8f acc[2][2] = {};

  auto issue = [&](int buf, int k0) {
#pragma unroll
    for (int i = 0; i < 4; ++i) {
      const int c = tid + i * 256;
      const int row = c >> 3, seg = (c & 7) * 8;
      async_cp16(&As[buf][row][seg],
                 A + (size_t)(m0 + row) * KH + k0 + seg);
    }
#pragma unroll
    for (int i = 0; i < 2; ++i) {
      const int c = tid + i * 256;
      const int n = c >> 3, seg = (c & 7) * 8;
      async_cp16(&Bs[buf][n][seg],
                 Dw + (size_t)(n0 + n) * KH + k0 + seg);
    }
  };

  issue(0, 0);
  int cur = 0;
  for (int t = 0; t < NKT; ++t) {
    if (t + 1 < NKT) {
      issue(cur ^ 1, (t + 1) * BK);
      WAIT_ASYNCCNT(6);
    } else {
      WAIT_ASYNCCNT(0);
    }
    __syncthreads();

#pragma unroll
    for (int kk = 0; kk < 2; ++kk) {
      v16bf af[2];
#pragma unroll
      for (int ms = 0; ms < 2; ++ms)
        af[ms] = load_a_frag(&As[cur][wm * 32 + ms * 16 + l16][kk * 32], half);
      v16bf bfr[2];
#pragma unroll
      for (int ns = 0; ns < 2; ++ns)
        bfr[ns] = load_b_frag(&Bs[cur][wn * 32 + ns * 16 + l16][kk * 32], half);
#pragma unroll
      for (int ms = 0; ms < 2; ++ms)
#pragma unroll
        for (int ns = 0; ns < 2; ++ns)
          acc[ms][ns] = __builtin_amdgcn_wmma_f32_16x16x32_bf16(
              false, af[ms], false, bfr[ns], (short)0, acc[ms][ns], false,
              false);
    }
    __syncthreads();
    cur ^= 1;
  }

#pragma unroll
  for (int ms = 0; ms < 2; ++ms)
#pragma unroll
    for (int ns = 0; ns < 2; ++ns)
#pragma unroll
      for (int r = 0; r < 8; ++r) {
        const int row = m0 + wm * 32 + ms * 16 + half * 8 + r;
        const int col = n0 + wn * 32 + ns * 16 + l16;
        Out[(size_t)row * KH + col] = f2bf(acc[ms][ns][r]);
      }
}

// ------------------------------- launcher -----------------------------------

extern "C" void kernel_launch(void* const* d_in, const int* in_sizes, int n_in,
                              void* d_out, int out_size, void* d_ws, size_t ws_size,
                              hipStream_t stream) {
  (void)in_sizes; (void)n_in; (void)out_size; (void)ws_size;
  const float* hidden = (const float*)d_in[0];
  const float* gate   = (const float*)d_in[1];
  const float* up     = (const float*)d_in[2];
  const float* down   = (const float*)d_in[3];

  char* ws = (char*)d_ws;
  const size_t SZ_A = (size_t)MTOT * KH * 2;   // 64 MB
  const size_t SZ_W = (size_t)KH * KH * 2;     // 32 MB each
  ushort_t* Abf   = (ushort_t*)(ws);
  ushort_t* GwT   = (ushort_t*)(ws + SZ_A);
  ushort_t* UwT   = (ushort_t*)(ws + SZ_A + SZ_W);
  ushort_t* DwT   = (ushort_t*)(ws + SZ_A + 2 * SZ_W);
  ushort_t* Gated = (ushort_t*)(ws + SZ_A + 3 * SZ_W);

  // hidden f32 -> bf16 : 33,554,432 elems / (256 thr * 8) = 16384 blocks exact
  cvt_f32_to_bf16<<<16384, 256, 0, stream>>>(hidden, Abf);

  // weights f32 -> bf16 transposed [N][K]
  cvtT_f32_to_bf16<<<dim3(KH / 32, KH / 32, 3), dim3(32, 8), 0, stream>>>(
      gate, up, down, GwT, UwT, DwT);

  // fused gate/up GEMM + GELU  (grid: N/64 x M/128)
  mlp_gateup<<<dim3(KH / 64, MTOT / 128), 256, 0, stream>>>(Abf, GwT, UwT, Gated);

  // down projection
  mlp_down<<<dim3(KH / 64, MTOT / 128), 256, 0, stream>>>(Gated, DwT,
                                                          (ushort_t*)d_out);
}
